// K2Layer_34497177321811
// MI455X (gfx1250) — matrix operand order repaired and checked by reading, not compiled, not run on hardware
//
#include <hip/hip_runtime.h>
#include <stdint.h>

// ---------------------------------------------------------------------------
// K2Layer for MI455X (gfx1250): bf16 WMMA everywhere, wave32, fp32 accumulate.
// B=8, W=1024, D=1024, R=64.
// ---------------------------------------------------------------------------

typedef __attribute__((ext_vector_type(16))) __bf16 v16bf;
typedef __attribute__((ext_vector_type(8)))  float  v8f;

union FragB { v16bf v; uint32_t u[8]; uint4 q[2]; };

static __device__ __forceinline__ uint16_t f2bf(float f) {
  __bf16 b = (__bf16)f;                       // hardware cvt, RNE
  return __builtin_bit_cast(uint16_t, b);
}
static __device__ __forceinline__ float sigm(float x) { return 1.0f / (1.0f + expf(-x)); }

#define BB 8
#define WW 1024
#define DD 1024
#define RR 64

// ---------------------------------------------------------------------------
// fp32 -> bf16 conversion (weights)
// ---------------------------------------------------------------------------
__global__ void k_cvt(const float* __restrict__ s, uint16_t* __restrict__ d, int n) {
  int i = blockIdx.x * blockDim.x + threadIdx.x;
  int stride = gridDim.x * blockDim.x;
  for (; i < n; i += stride) d[i] = f2bf(s[i]);
}

// uvT[r][k] = concat(u,v)[k][r], bf16, [2R][D] row-major (B-operand layout)
__global__ void k_pack_uvT(const float* __restrict__ u, const float* __restrict__ v,
                           uint16_t* __restrict__ dst) {
  int i = blockIdx.x * blockDim.x + threadIdx.x;
  int stride = gridDim.x * blockDim.x;
  int n = 2 * RR * DD;
  for (; i < n; i += stride) {
    int r = i >> 10, k = i & (DD - 1);
    float val = (r < RR) ? u[k * RR + r] : v[k * RR + (r - RR)];
    dst[i] = f2bf(val);
  }
}

// ---------------------------------------------------------------------------
// RMSNorm row kernel: y_bf16 = x * rsqrt(mean(x^2)+1e-8) * scale
// ---------------------------------------------------------------------------
__global__ __launch_bounds__(256) void k_rmsnorm(const float* __restrict__ x,
                                                 const float* __restrict__ scale,
                                                 uint16_t* __restrict__ y) {
  __shared__ float red[256];
  const int row = blockIdx.x, t = threadIdx.x;
  float4 vv = ((const float4*)(x + (size_t)row * DD))[t];
  red[t] = vv.x * vv.x + vv.y * vv.y + vv.z * vv.z + vv.w * vv.w;
  __syncthreads();
  for (int s = 128; s > 0; s >>= 1) {
    if (t < s) red[t] += red[t + s];
    __syncthreads();
  }
  float inv = rsqrtf(red[0] * (1.0f / DD) + 1e-8f);
  uint16_t* yr = y + (size_t)row * DD;
  int c = t * 4;
  yr[c + 0] = f2bf(vv.x * inv * scale[c + 0]);
  yr[c + 1] = f2bf(vv.y * inv * scale[c + 1]);
  yr[c + 2] = f2bf(vv.z * inv * scale[c + 2]);
  yr[c + 3] = f2bf(vv.w * inv * scale[c + 3]);
}

// ---------------------------------------------------------------------------
// One-shot transpose of hn into K-major panels:
// hnT[((b*32 + c)*1024 + col)*32 + row] = hn[(b*1024 + c*32 + row)*1024 + col]
// ---------------------------------------------------------------------------
__global__ __launch_bounds__(256) void k_transpose(const uint16_t* __restrict__ hn,
                                                   uint16_t* __restrict__ hnT) {
  __shared__ uint16_t tile[32][72];  // padded
  const int t = threadIdx.x;
  const int bc = blockIdx.y;            // b*32 + c
  const int colbase = blockIdx.x * 64;
  {
    int row = t >> 3, c8 = (t & 7) * 8;
    uint4 vls = *(const uint4*)(hn + (size_t)(bc * 32 + row) * DD + colbase + c8);
    const uint16_t* e = (const uint16_t*)&vls;
#pragma unroll
    for (int k = 0; k < 8; ++k) tile[row][c8 + k] = e[k];
  }
  __syncthreads();
  {
    int col = t >> 2, rgrp = (t & 3) * 8;
    uint4 o;
    uint32_t* ow = (uint32_t*)&o;
#pragma unroll
    for (int k = 0; k < 4; ++k)
      ow[k] = (uint32_t)tile[rgrp + 2 * k][col] | ((uint32_t)tile[rgrp + 2 * k + 1][col] << 16);
    *(uint4*)(hnT + ((size_t)bc * DD + colbase + col) * 32 + rgrp) = o;
  }
}

// ---------------------------------------------------------------------------
// q/k projection as WMMA GEMM: qkraw[M=8192][128] = hn @ uvT^T (fp32 raw)
// block = 256 (8 waves: 2 M-groups x 4 N-groups), wave = 4 M-tiles x 2 N-tiles,
// register double-buffered over K.
// ---------------------------------------------------------------------------
__global__ __launch_bounds__(256) void k_qkproj(const uint16_t* __restrict__ hn,
                                                const uint16_t* __restrict__ uvT,
                                                float* __restrict__ qkraw) {
  const int t = threadIdx.x, wave = t >> 5, lane = t & 31;
  const int li = lane & 15, half = lane >> 4;
  const int rowBase = blockIdx.x * 128 + (wave >> 2) * 64;
  const int colBase = (wave & 3) * 32;
  const int koffA = half ? 16 : 0, koffB = half ? 16 : 0;

  v8f acc[4][2];
#pragma unroll
  for (int mi = 0; mi < 4; ++mi)
#pragma unroll
    for (int ni = 0; ni < 2; ++ni) acc[mi][ni] = (v8f){0, 0, 0, 0, 0, 0, 0, 0};

  auto loadAB = [&](FragB (&aa)[4], FragB (&bb)[2], int ks) {
#pragma unroll
    for (int mi = 0; mi < 4; ++mi) {
      const uint16_t* ar = hn + (size_t)(rowBase + mi * 16 + li) * DD + ks + koffA;
      aa[mi].q[0] = *(const uint4*)ar;
      aa[mi].q[1] = *(const uint4*)(ar + 8);
    }
#pragma unroll
    for (int ni = 0; ni < 2; ++ni) {
      const uint16_t* br = uvT + (size_t)(colBase + ni * 16 + li) * DD + ks + koffB;
      bb[ni].q[0] = *(const uint4*)br;
      bb[ni].q[1] = *(const uint4*)(br + 8);
    }
  };
  auto mmac = [&](FragB (&aa)[4], FragB (&bb)[2]) {
#pragma unroll
    for (int mi = 0; mi < 4; ++mi)
#pragma unroll
      for (int ni = 0; ni < 2; ++ni)
        acc[mi][ni] = __builtin_amdgcn_wmma_f32_16x16x32_bf16(
            false, aa[mi].v, false, bb[ni].v, (short)0, acc[mi][ni], false, false);
  };

  FragB a0[4], b0[2], a1[4], b1[2];
  loadAB(a0, b0, 0);
  int ks = 0;
  for (; ks + 64 < DD; ks += 64) {
    loadAB(a1, b1, ks + 32);
    mmac(a0, b0);
    loadAB(a0, b0, ks + 64);
    mmac(a1, b1);
  }
  loadAB(a1, b1, ks + 32);
  mmac(a0, b0);
  mmac(a1, b1);

#pragma unroll
  for (int mi = 0; mi < 4; ++mi)
#pragma unroll
    for (int ni = 0; ni < 2; ++ni)
#pragma unroll
      for (int g = 0; g < 8; ++g)
        qkraw[(size_t)(rowBase + mi * 16 + g + 8 * half) * 128 + colBase + ni * 16 + li] =
            acc[mi][ni][g];
}

// ---------------------------------------------------------------------------
// L2-norm + gamma pre-scale of raw q/k.
// ---------------------------------------------------------------------------
__global__ __launch_bounds__(128) void k_qknorm(const float* __restrict__ qkraw,
                                                const float* __restrict__ decay_logit,
                                                float* __restrict__ qpre,
                                                uint16_t* __restrict__ kpre) {
  __shared__ float red[128];
  const int idx = blockIdx.x, t = threadIdx.x;
  float acc = qkraw[(size_t)idx * 128 + t];
  red[t] = acc * acc;
  __syncthreads();
  for (int s = 32; s > 0; s >>= 1) {
    if ((t & 63) < s) red[t] += red[t + s];
    __syncthreads();
  }
  float nrm = sqrtf(red[t & 64]);
  float val = acc / fmaxf(nrm, 1e-8f);
  const int i = idx & (WW - 1);
  const float ph = (float)(i & 31);
  const int r = t & 63;
  float g = 0.15f + 0.85f * sigm(decay_logit[r]);
  float lg = log2f(fmaxf(g, 1e-8f));
  if (t < RR)
    qpre[(size_t)idx * RR + r] = val * exp2f(ph * lg);
  else
    kpre[(size_t)idx * RR + r] = f2bf(val * exp2f(-ph * lg));
}

// ---------------------------------------------------------------------------
// Mixing kernel (flash style), grid = (64 i-tiles, 8 batch), block = 512 (16 waves).
// Dynamic LDS (130 KB, legal on gfx1250's 320KB/WG): two full-chunk ping-pong
// panels -> ONE barrier per 32-wide j-chunk; per stage each wave runs 4 WMMAs
// while the next chunk's 64KB panel streams into the other buffer and waves 0,1
// produce the next W tile (score WMMAs + gate*k_base + causal mask).
// Buffer addresses are computed inline (no pointer aggregates -> no static
// addrspacecast initializers).
// ---------------------------------------------------------------------------
__global__ __launch_bounds__(512) void k_mix(const uint16_t* __restrict__ hnT,
                                             const float* __restrict__ kbase,
                                             const float* __restrict__ qpre,
                                             const uint16_t* __restrict__ kpre,
                                             const float* __restrict__ decay_logit,
                                             const float* __restrict__ gate_logit,
                                             const float* __restrict__ alpha_logit,
                                             uint16_t* __restrict__ outb) {
  extern __shared__ uint16_t dynlds[];  // [0,65536): two 32K-half panels; then 2x512 W tiles

  const int t = threadIdx.x, wave = t >> 5, lane = t & 31;
  const int li = lane & 15, half = lane >> 4;
  const int it = blockIdx.x, b = blockIdx.y;
  const int i0 = it * 16, m0 = it >> 1;
  const int koffA = half ? 16 : 0;
  const int koffB = half * 16;

  const float gatev = sigm(gate_logit[0]);
  const float alphav = sigm(alpha_logit[0]);  // ALPHA_CAP = 1.0

  float qA[32], scl[32], g32[32];
  if (wave < 2) {
    const float* qrow = qpre + (size_t)(b * WW + i0 + li) * RR;
#pragma unroll
    for (int f = 0; f < 2; ++f)
#pragma unroll
      for (int s = 0; s < 16; ++s) {
        int r = f * 32 + koffA + s;
        qA[f * 16 + s] = qrow[r];
        float g = 0.15f + 0.85f * sigm(decay_logit[r]);
        g32[f * 16 + s] = exp2f(32.0f * log2f(fmaxf(g, 1e-8f)));
        scl[f * 16 + s] = 1.0f;
      }
  }

  // scores for chunk cc -> W buffer (cc&1); caller guarantees wave<2
  auto do_scores = [&](int cc) {
    const int jcol = cc * 32 + wave * 16 + li;
    const uint16_t* kprow = kpre + (size_t)(b * WW + jcol) * RR;
    FragB kf0, kf1, qf0, qf1;
    kf0.q[0] = *(const uint4*)(kprow + koffB);
    kf0.q[1] = *(const uint4*)(kprow + koffB + 8);
    kf1.q[0] = *(const uint4*)(kprow + 32 + koffB);
    kf1.q[1] = *(const uint4*)(kprow + 32 + koffB + 8);
#pragma unroll
    for (int g = 0; g < 8; ++g) {
      qf0.u[g] = (uint32_t)f2bf(qA[2 * g] * scl[2 * g]) |
                 ((uint32_t)f2bf(qA[2 * g + 1] * scl[2 * g + 1]) << 16);
      qf1.u[g] = (uint32_t)f2bf(qA[16 + 2 * g] * scl[16 + 2 * g]) |
                 ((uint32_t)f2bf(qA[16 + 2 * g + 1] * scl[16 + 2 * g + 1]) << 16);
    }
    v8f sc = (v8f){0, 0, 0, 0, 0, 0, 0, 0};
    sc = __builtin_amdgcn_wmma_f32_16x16x32_bf16(false, qf0.v, false, kf0.v, (short)0, sc, false, false);
    sc = __builtin_amdgcn_wmma_f32_16x16x32_bf16(false, qf1.v, false, kf1.v, (short)0, sc, false, false);
    uint16_t* wdst = dynlds + 64 * 1024 + (cc & 1) * 512;
#pragma unroll
    for (int g = 0; g < 8; ++g) {
      int irow = i0 + g + 8 * half;
      float sv = (jcol <= irow) ? sc[g] : 0.0f;
      float wv = gatev * kbase[(size_t)irow * WW + jcol] + alphav * sv;
      wdst[(g + 8 * half) * 32 + (wave * 16 + li)] = f2bf(wv);
    }
#pragma unroll
    for (int x = 0; x < 32; ++x) scl[x] *= g32[x];
  };

  v8f acc[4];
#pragma unroll
  for (int n = 0; n < 4; ++n) acc[n] = (v8f){0, 0, 0, 0, 0, 0, 0, 0};

  // prologue: stream chunk m0's full panel + its W tile
  {
    const uint4* s4 = (const uint4*)(hnT + (size_t)(b * 32 + m0) * DD * 32);
    uint4* d4 = (uint4*)dynlds;
#pragma unroll
    for (int k2 = 0; k2 < 8; ++k2) d4[t + 512 * k2] = s4[t + 512 * k2];
    if (wave < 2) do_scores(m0);
  }

  for (int s = 0; s <= m0; ++s) {
    __syncthreads();
    const int c = m0 - s;

    // overlap: stream-copy next chunk's 64KB panel into the other buffer
    if (c > 0) {
      const uint4* s4 = (const uint4*)(hnT + (size_t)(b * 32 + c - 1) * DD * 32);
      uint4* d4 = (uint4*)(dynlds + ((s + 1) & 1) * (32 * 1024));
#pragma unroll
      for (int k2 = 0; k2 < 8; ++k2) d4[t + 512 * k2] = s4[t + 512 * k2];
    }
    // overlap: prefetch chunk c-2 (global_prefetch_b8)
    if (c > 1)
      __builtin_prefetch(
          (const char*)(hnT + (size_t)(b * 32 + c - 2) * DD * 32) + t * 128, 0, 0);
    // overlap: scores for the next chunk
    if (c > 0 && wave < 2) do_scores(c - 1);

    // compute: 4 WMMAs per wave from the buffers filled last stage
    FragB wf;
    const uint16_t* wbuf = dynlds + 64 * 1024 + (c & 1) * 512;
    wf.q[0] = *(const uint4*)&wbuf[li * 32 + koffA];
    wf.q[1] = *(const uint4*)&wbuf[li * 32 + koffA + 8];
    const uint16_t* hbuf = dynlds + (s & 1) * (32 * 1024);
#pragma unroll
    for (int nn = 0; nn < 4; ++nn) {
      const int colloc = (wave * 4 + nn) * 16 + li;
      FragB hf;
      hf.q[0] = *(const uint4*)&hbuf[colloc * 32 + koffB];
      hf.q[1] = *(const uint4*)&hbuf[colloc * 32 + koffB + 8];
      acc[nn] = __builtin_amdgcn_wmma_f32_16x16x32_bf16(false, wf.v, false, hf.v,
                                                        (short)0, acc[nn], false, false);
    }
  }

  // --- store mix output (bf16) ---
#pragma unroll
  for (int nn = 0; nn < 4; ++nn) {
    int colbase = (wave * 4 + nn) * 16 + li;
#pragma unroll
    for (int g = 0; g < 8; ++g) {
      int row = i0 + g + 8 * half;
      outb[(size_t)(b * WW + row) * DD + colbase] = f2bf(acc[nn][g]);
    }
  }
}

// ---------------------------------------------------------------------------
// Generic bf16 WMMA GEMM: out = act( A[M,K] @ Wt[N,K]^T + bias[N] (+ res[M,N]) )
// block = 256 (8 waves, 2 M x 4 N), wave computes 32x64; block tile 64x256.
// Register double-buffered over K so loads overlap WMMA.
// ---------------------------------------------------------------------------
template <int OUT_BF16, int DO_GELU, int HAS_RES>
__global__ __launch_bounds__(256) void k_gemm(const uint16_t* __restrict__ A,
                                              const uint16_t* __restrict__ Wt,
                                              const float* __restrict__ bias,
                                              const float* __restrict__ res,
                                              void* __restrict__ out,
                                              int M, int N, int K) {
  const int t = threadIdx.x, wave = t >> 5, lane = t & 31;
  const int li = lane & 15, half = lane >> 4;
  const int rowBase = blockIdx.y * 64 + (wave >> 2) * 32;
  const int colBase = blockIdx.x * 256 + (wave & 3) * 64;
  const int koffA = half ? 16 : 0;
  const int koffB = half ? 16 : 0;

  v8f acc[2][4];
#pragma unroll
  for (int mi = 0; mi < 2; ++mi)
#pragma unroll
    for (int ni = 0; ni < 4; ++ni) acc[mi][ni] = (v8f){0, 0, 0, 0, 0, 0, 0, 0};

  auto loadAB = [&](FragB (&aa)[2], FragB (&bb)[4], int ks) {
#pragma unroll
    for (int mi = 0; mi < 2; ++mi) {
      const uint16_t* ar = A + (size_t)(rowBase + mi * 16 + li) * K + ks + koffA;
      aa[mi].q[0] = *(const uint4*)ar;
      aa[mi].q[1] = *(const uint4*)(ar + 8);
    }
#pragma unroll
    for (int ni = 0; ni < 4; ++ni) {
      const uint16_t* br = Wt + (size_t)(colBase + ni * 16 + li) * K + ks + koffB;
      bb[ni].q[0] = *(const uint4*)br;
      bb[ni].q[1] = *(const uint4*)(br + 8);
    }
  };
  auto mmac = [&](FragB (&aa)[2], FragB (&bb)[4]) {
#pragma unroll
    for (int mi = 0; mi < 2; ++mi)
#pragma unroll
      for (int ni = 0; ni < 4; ++ni)
        acc[mi][ni] = __builtin_amdgcn_wmma_f32_16x16x32_bf16(
            false, aa[mi].v, false, bb[ni].v, (short)0, acc[mi][ni], false, false);
  };

  FragB a0[2], b0[4], a1[2], b1[4];
  loadAB(a0, b0, 0);
  int ks = 0;
  for (; ks + 64 < K; ks += 64) {    // K is a multiple of 64
    loadAB(a1, b1, ks + 32);
    mmac(a0, b0);
    loadAB(a0, b0, ks + 64);
    mmac(a1, b1);
  }
  loadAB(a1, b1, ks + 32);
  mmac(a0, b0);
  mmac(a1, b1);

#pragma unroll
  for (int mi = 0; mi < 2; ++mi)
#pragma unroll
    for (int ni = 0; ni < 4; ++ni) {
      int col = colBase + ni * 16 + li;
      float bs = bias[col];
#pragma unroll
      for (int g = 0; g < 8; ++g) {
        int row = rowBase + mi * 16 + g + 8 * half;
        float v = acc[mi][ni][g] + bs;
        if (HAS_RES) v += res[(size_t)row * N + col];
        if (DO_GELU) v = 0.5f * v * (1.0f + erff(v * 0.70710678118654752f));
        if (OUT_BF16)
          ((uint16_t*)out)[(size_t)row * N + col] = f2bf(v);
        else
          ((float*)out)[(size_t)row * N + col] = v;
      }
    }
}

// ---------------------------------------------------------------------------
// Launch
// ---------------------------------------------------------------------------
extern "C" void kernel_launch(void* const* d_in, const int* in_sizes, int n_in,
                              void* d_out, int out_size, void* d_ws, size_t ws_size,
                              hipStream_t stream) {
  const float* h           = (const float*)d_in[0];
  const float* k_base      = (const float*)d_in[1];
  const float* decay_logit = (const float*)d_in[2];
  const float* gate_logit  = (const float*)d_in[3];
  const float* alpha_logit = (const float*)d_in[4];
  const float* u           = (const float*)d_in[5];
  const float* v           = (const float*)d_in[6];
  const float* proj_w      = (const float*)d_in[7];
  const float* proj_b      = (const float*)d_in[8];
  const float* norm1       = (const float*)d_in[9];
  const float* norm2       = (const float*)d_in[10];
  const float* up_w        = (const float*)d_in[11];
  const float* up_b        = (const float*)d_in[12];
  const float* down_w      = (const float*)d_in[13];
  const float* down_b      = (const float*)d_in[14];

  char* ws = (char*)d_ws;
  size_t off = 0;
  auto alloc = [&](size_t bytes) -> void* {
    void* p = ws + off;
    off += (bytes + 255) & ~(size_t)255;
    return p;
  };
  uint16_t* hn    = (uint16_t*)alloc((size_t)BB * WW * DD * 2);       // 16 MB
  uint16_t* hnT   = (uint16_t*)alloc((size_t)BB * WW * DD * 2);       // 16 MB
  uint16_t* mix   = (uint16_t*)alloc((size_t)BB * WW * DD * 2);       // 16 MB
  float*    h1    = (float*)   alloc((size_t)BB * WW * DD * 4);       // 32 MB
  uint16_t* mn    = (uint16_t*)alloc((size_t)BB * WW * DD * 2);       // 16 MB
  uint16_t* act   = (uint16_t*)alloc((size_t)BB * WW * 2 * DD * 2);   // 32 MB
  uint16_t* uvT   = (uint16_t*)alloc((size_t)2 * RR * DD * 2);
  uint16_t* pwb   = (uint16_t*)alloc((size_t)DD * DD * 2);
  uint16_t* uwb   = (uint16_t*)alloc((size_t)2 * DD * DD * 2);
  uint16_t* dwb   = (uint16_t*)alloc((size_t)DD * 2 * DD * 2);
  float*    qkraw = (float*)   alloc((size_t)BB * WW * 128 * 4);      // 4 MB
  float*    qpre  = (float*)   alloc((size_t)BB * WW * RR * 4);
  uint16_t* kpre  = (uint16_t*)alloc((size_t)BB * WW * RR * 2);

  // weight conversions
  k_cvt<<<512, 256, 0, stream>>>(proj_w, pwb, DD * DD);
  k_cvt<<<512, 256, 0, stream>>>(up_w, uwb, 2 * DD * DD);
  k_cvt<<<512, 256, 0, stream>>>(down_w, dwb, DD * 2 * DD);
  k_pack_uvT<<<256, 256, 0, stream>>>(u, v, uvT);

  // h_norm (bf16)
  k_rmsnorm<<<BB * WW, 256, 0, stream>>>(h, norm1, hn);

  // K-major 32-row panels for the mixing kernel
  k_transpose<<<dim3(DD / 64, BB * 32), 256, 0, stream>>>(hn, hnT);

  // q/k projection (WMMA) + l2norm + gamma pre-scale
  k_qkproj<<<BB * WW / 128, 256, 0, stream>>>(hn, uvT, qkraw);
  k_qknorm<<<BB * WW, 128, 0, stream>>>(qkraw, decay_logit, qpre, kpre);

  // mixing: gate*kb@hn + alpha*scores@hn  -> bf16  (130KB dynamic LDS)
  const size_t mix_lds = (size_t)(2 * 32 * 1024 + 2 * 512) * sizeof(uint16_t);
  k_mix<<<dim3(WW / 16, BB), 512, mix_lds, stream>>>(hnT, k_base, qpre, kpre, decay_logit,
                                                     gate_logit, alpha_logit, mix);

  // h1 = h + mix @ proj_w^T + proj_b   (fp32)
  k_gemm<0, 0, 1><<<dim3(DD / 256, BB * WW / 64), 256, 0, stream>>>(
      mix, pwb, proj_b, h, (void*)h1, BB * WW, DD, DD);

  // m = rmsnorm(h1)  (bf16)
  k_rmsnorm<<<BB * WW, 256, 0, stream>>>(h1, norm2, mn);

  // act = gelu(m @ up_w^T + up_b)  (bf16)
  k_gemm<1, 1, 0><<<dim3(2 * DD / 256, BB * WW / 64), 256, 0, stream>>>(
      mn, uwb, up_b, nullptr, (void*)act, BB * WW, 2 * DD, DD);

  // out = h1 + act @ down_w^T + down_b  (fp32)
  k_gemm<0, 0, 1><<<dim3(DD / 256, BB * WW / 64), 256, 0, stream>>>(
      act, dwb, down_b, h1, d_out, BB * WW, DD, 2 * DD);
}
